// MultiHeadAttention_16003048145572
// MI455X (gfx1250) — matrix operand order repaired and checked by reading, not compiled, not run on hardware
//
#include <hip/hip_runtime.h>
#include <hip/hip_bf16.h>

// MI455X (gfx1250) flash-attention + fused output projection, wave32 bf16 WMMA.
// Round 4: same structure as round 3 (32 Q-rows/wave, shared B operands,
// 64-key chunks), plus amdgpu_waves_per_eu(4) on the attention kernel to cap
// the allocator at 256 VGPRs: round 3's schedule spilled into the v256+ range
// and paid s_set_vgpr_msb churn in the softmax block.

#define S_LEN 4096
#define EMBED 512
#define HEADS 8
#define HDIM  64
#define KCH   64      // keys per chunk
#define WAVES 4       // waves per block (128 threads)

typedef __attribute__((ext_vector_type(16))) __bf16 bf16x16;
typedef __attribute__((ext_vector_type(8)))  __bf16 bf16x8;
typedef __attribute__((ext_vector_type(4)))  __bf16 bf16x4;
typedef __attribute__((ext_vector_type(8)))  float  f32x8;

#define WMMA_BF16(A, B, C) \
  __builtin_amdgcn_wmma_f32_16x16x32_bf16(false, (A), false, (B), (short)0, (C), false, false)

union F4 { float4 v; float f[4]; };

__device__ __forceinline__ bf16x16 cat16(bf16x8 lo, bf16x8 hi) {
  return __builtin_shufflevector(lo, hi, 0, 1, 2, 3, 4, 5, 6, 7,
                                 8, 9, 10, 11, 12, 13, 14, 15);
}
// B operand: lane = N column, elements = K rows [base..base+7],[base+16..base+23] (row stride 64)
__device__ __forceinline__ bf16x16 ldsB(const __bf16* p) {
  bf16x16 r;
#pragma unroll
  for (int s = 0; s < 8; ++s) { r[s] = p[s * 64]; r[8 + s] = p[(16 + s) * 64]; }
  return r;
}
// A operand (ISA 7.12.2): per-lane K runs [base..base+7],[base+16..base+23]
__device__ __forceinline__ bf16x16 ldsA(const __bf16* p) {
  return cat16(*(const bf16x8*)p, *(const bf16x8*)(p + 16));
}
// 8 f32 from global -> bf16x8 (two global_load_b128 + packed converts)
__device__ __forceinline__ bf16x8 cvt8(const float* p) {
  F4 a, b;
  a.v = *(const float4*)p;
  b.v = *(const float4*)(p + 4);
  bf16x8 r;
#pragma unroll
  for (int i = 0; i < 4; ++i) { r[i] = (__bf16)a.f[i]; r[4 + i] = (__bf16)b.f[i]; }
  return r;
}

template <int XORMASK>
__device__ __forceinline__ float swz_xor(float v) {
  int r = __builtin_amdgcn_ds_swizzle(__builtin_bit_cast(int, v),
                                      (XORMASK << 10) | 0x1F);
  return __builtin_bit_cast(float, r);
}
__device__ __forceinline__ float rowmax16(float v) {
  v = fmaxf(v, swz_xor<1>(v));
  v = fmaxf(v, swz_xor<2>(v));
  v = fmaxf(v, swz_xor<4>(v));
  v = fmaxf(v, swz_xor<8>(v));
  return v;
}
__device__ __forceinline__ float rowsum16(float v) {
  v += swz_xor<1>(v);
  v += swz_xor<2>(v);
  v += swz_xor<4>(v);
  v += swz_xor<8>(v);
  return v;
}

// online-softmax update for one 16x64 score tile set; rescales 4 O tiles and
// writes the P tile (bf16) into this wave/row-tile's LDS relayout buffer
__device__ __forceinline__ void softmax_update(f32x8* s, float* m, float* l,
                                               f32x8* o4, __bf16 (*Prow)[KCH],
                                               int l16, int half, float sc2) {
  float corr[8];
#pragma unroll
  for (int j = 0; j < 8; ++j) {
    const float a0 = s[0][j] * sc2, a1 = s[1][j] * sc2;
    const float a2 = s[2][j] * sc2, a3 = s[3][j] * sc2;
    float mx = rowmax16(fmaxf(fmaxf(a0, a1), fmaxf(a2, a3)));
    const float mn = fmaxf(m[j], mx);
    corr[j] = exp2f(m[j] - mn);
    const float p0 = exp2f(a0 - mn), p1 = exp2f(a1 - mn);
    const float p2 = exp2f(a2 - mn), p3 = exp2f(a3 - mn);
    l[j] = l[j] * corr[j] + rowsum16((p0 + p1) + (p2 + p3));
    m[j] = mn;
    s[0][j] = p0; s[1][j] = p1; s[2][j] = p2; s[3][j] = p3;
  }
#pragma unroll
  for (int dt = 0; dt < 4; ++dt)
#pragma unroll
    for (int j = 0; j < 8; ++j) o4[dt][j] *= corr[j];
#pragma unroll
  for (int j = 0; j < 8; ++j) {
    const int row = j + 8 * half;
    Prow[row][l16]      = (__bf16)s[0][j];
    Prow[row][16 + l16] = (__bf16)s[1][j];
    Prow[row][32 + l16] = (__bf16)s[2][j];
    Prow[row][48 + l16] = (__bf16)s[3][j];
  }
}

__global__ __launch_bounds__(128)
__attribute__((amdgpu_waves_per_eu(4)))
void attn_flash_kernel(
    const float* __restrict__ Vg, const float* __restrict__ Kg,
    const float* __restrict__ Qg, float* __restrict__ Og) {
  __shared__ alignas(16) __bf16 KldsT[HDIM][KCH];          // [d][key]   8KB
  __shared__ alignas(16) __bf16 Vlds[KCH][HDIM];           // [key][d]   8KB
  __shared__ alignas(16) __bf16 Plds[WAVES][2][16][KCH];   // P relayout 16KB

  const int tid  = threadIdx.x;
  const int wave = tid >> 5;
  const int lane = tid & 31;
  const int l16  = lane & 15;
  const int half = lane >> 4;

  const int qblk = blockIdx.x;   // 128 q-rows per block, 32 per wave
  const int h    = blockIdx.y;
  const int n    = blockIdx.z;

  const int    qrow0  = qblk * 128 + wave * 32;
  const size_t baseNH = (size_t)n * S_LEN * EMBED + (size_t)h * HDIM;

  // ---- Q (32x64) -> four A operands (2 row tiles x 2 d blocks) ----
  bf16x16 qa00, qa01, qa10, qa11;
  {
    const float* q0 = Qg + baseNH + (size_t)(qrow0 + l16) * EMBED + half * 8;
    const float* q1 = q0 + 16 * EMBED;
    qa00 = cat16(cvt8(q0), cvt8(q0 + 16));
    qa01 = cat16(cvt8(q0 + 32), cvt8(q0 + 48));
    qa10 = cat16(cvt8(q1), cvt8(q1 + 16));
    qa11 = cat16(cvt8(q1 + 32), cvt8(q1 + 48));
  }

  float m0[8], l0[8], m1[8], l1[8];
#pragma unroll
  for (int j = 0; j < 8; ++j) {
    m0[j] = -3.0e38f; l0[j] = 0.0f;
    m1[j] = -3.0e38f; l1[j] = 0.0f;
  }
  f32x8 o[8] = {{}, {}, {}, {}, {}, {}, {}, {}};   // two 16x64 f32 tile sets

  const float sc2 = 0.044194173824159216f * 1.4426950408889634f;

  for (int kb = 0; kb < S_LEN; kb += KCH) {
    // ---- stage chunk: K transposed (4x4 register sub-tiles), V natural ----
#pragma unroll
    for (int t = 0; t < 2; ++t) {
      const int T    = tid + t * 128;   // 256 tiles of 4key x 4d
      const int key0 = (T & 15) * 4;
      const int d0   = (T >> 4) * 4;
      F4 kr[4], vr[4];
#pragma unroll
      for (int r = 0; r < 4; ++r) {
        const size_t g = baseNH + (size_t)(kb + key0 + r) * EMBED + d0;
        kr[r].v = *(const float4*)(Kg + g);
        vr[r].v = *(const float4*)(Vg + g);
      }
#pragma unroll
      for (int i = 0; i < 4; ++i) {
        bf16x4 kk;
#pragma unroll
        for (int r = 0; r < 4; ++r) kk[r] = (__bf16)kr[r].f[i];
        *(bf16x4*)&KldsT[d0 + i][key0] = kk;
      }
#pragma unroll
      for (int r = 0; r < 4; ++r) {
        bf16x4 vv;
#pragma unroll
        for (int i = 0; i < 4; ++i) vv[i] = (__bf16)vr[r].f[i];
        *(bf16x4*)&Vlds[key0 + r][d0] = vv;
      }
    }
    __syncthreads();

    // prefetch next chunk (global_prefetch_b8)
    if (kb + KCH < S_LEN) {
      const size_t pr = baseNH + (size_t)(kb + KCH + (tid >> 1)) * EMBED + (tid & 1) * 32;
      __builtin_prefetch(Kg + pr, 0, 1);
      __builtin_prefetch(Vg + pr, 0, 1);
    }

    // ---- S = Q*K^T : two 16x64 tiles share each B pair (16 WMMAs) ----
    f32x8 s0[4], s1[4];
#pragma unroll
    for (int nt = 0; nt < 4; ++nt) {
      const bf16x16 b0 = ldsB(&KldsT[half * 8][nt * 16 + l16]);        // d 0..31
      const bf16x16 b1 = ldsB(&KldsT[32 + half * 8][nt * 16 + l16]);   // d 32..63
      const f32x8 z = {};
      s0[nt] = WMMA_BF16(qa00, b0, z);
      s0[nt] = WMMA_BF16(qa01, b1, s0[nt]);
      s1[nt] = WMMA_BF16(qa10, b0, z);
      s1[nt] = WMMA_BF16(qa11, b1, s1[nt]);
    }

    // ---- online softmax per row tile; P -> LDS relayout ----
    softmax_update(s0, m0, l0, &o[0], Plds[wave][0], l16, half, sc2);
    softmax_update(s1, m1, l1, &o[4], Plds[wave][1], l16, half, sc2);

    const bf16x16 pa00 = ldsA(&Plds[wave][0][l16][half * 8]);        // keys 0..31
    const bf16x16 pa01 = ldsA(&Plds[wave][0][l16][32 + half * 8]);   // keys 32..63
    const bf16x16 pa10 = ldsA(&Plds[wave][1][l16][half * 8]);
    const bf16x16 pa11 = ldsA(&Plds[wave][1][l16][32 + half * 8]);

    // ---- O += P*V : two row tiles share each B pair (16 WMMAs) ----
#pragma unroll
    for (int dt = 0; dt < 4; ++dt) {
      const bf16x16 bv0 = ldsB(&Vlds[half * 8][dt * 16 + l16]);        // keys 0..31
      const bf16x16 bv1 = ldsB(&Vlds[32 + half * 8][dt * 16 + l16]);   // keys 32..63
      o[dt]     = WMMA_BF16(pa00, bv0, o[dt]);
      o[dt]     = WMMA_BF16(pa01, bv1, o[dt]);
      o[4 + dt] = WMMA_BF16(pa10, bv0, o[4 + dt]);
      o[4 + dt] = WMMA_BF16(pa11, bv1, o[4 + dt]);
    }
    __syncthreads();
  }

  // ---- normalize + store both row tiles ----
#pragma unroll
  for (int rt = 0; rt < 2; ++rt) {
    const float* lp = rt ? l1 : l0;
#pragma unroll
    for (int j = 0; j < 8; ++j) {
      const float inv = 1.0f / lp[j];
      const int row = qrow0 + rt * 16 + j + 8 * half;
      float* op = Og + (size_t)n * S_LEN * EMBED + (size_t)row * EMBED +
                  (size_t)h * HDIM + l16;
      op[0]  = o[rt * 4 + 0][j] * inv;
      op[16] = o[rt * 4 + 1][j] * inv;
      op[32] = o[rt * 4 + 2][j] * inv;
      op[48] = o[rt * 4 + 3][j] * inv;
    }
  }
}

// Y[M, EMBED] = X[M, EMBED] @ W^T + b  (torch Linear), M = N*S
__global__ __launch_bounds__(128) void fc_out_kernel(
    const float* __restrict__ X, const float* __restrict__ W,
    const float* __restrict__ Bv, float* __restrict__ Y) {
  __shared__ alignas(16) __bf16 Alds[64][32];   // [m][k]
  __shared__ alignas(16) __bf16 Blds[32][64];   // [k][o] = W^T slice

  const int tid  = threadIdx.x;
  const int wave = tid >> 5;
  const int lane = tid & 31;
  const int l16  = lane & 15;
  const int half = lane >> 4;

  const int m0 = blockIdx.x * 64;
  const int o0 = blockIdx.y * 64;

  f32x8 acc[4] = {{}, {}, {}, {}};

  for (int kb = 0; kb < EMBED; kb += 32) {
#pragma unroll
    for (int i = 0; i < 4; ++i) {
      const int f4  = tid + i * 128;
      const int row = f4 >> 3;
      const int c   = (f4 & 7) * 4;
      F4 a; a.v = *(const float4*)(X + (size_t)(m0 + row) * EMBED + kb + c);
      bf16x4 t;
#pragma unroll
      for (int q = 0; q < 4; ++q) t[q] = (__bf16)a.f[q];
      *(bf16x4*)&Alds[row][c] = t;
    }
    {
      const int or0 = (tid & 15) * 4;
      const int k0  = (tid >> 4) * 4;
      F4 wr[4];
#pragma unroll
      for (int r = 0; r < 4; ++r)
        wr[r].v = *(const float4*)(W + (size_t)(o0 + or0 + r) * EMBED + kb + k0);
#pragma unroll
      for (int i = 0; i < 4; ++i) {
        bf16x4 t;
#pragma unroll
        for (int r = 0; r < 4; ++r) t[r] = (__bf16)wr[r].f[i];
        *(bf16x4*)&Blds[k0 + i][or0] = t;
      }
    }
    __syncthreads();

    const bf16x16 a = ldsA(&Alds[wave * 16 + l16][half * 8]);
#pragma unroll
    for (int ot = 0; ot < 4; ++ot) {
      const bf16x16 b = ldsB(&Blds[half * 8][ot * 16 + l16]);
      acc[ot] = WMMA_BF16(a, b, acc[ot]);
    }
    __syncthreads();
  }

  const float bb0 = Bv[o0 + l16];
  const float bb1 = Bv[o0 + 16 + l16];
  const float bb2 = Bv[o0 + 32 + l16];
  const float bb3 = Bv[o0 + 48 + l16];
#pragma unroll
  for (int j = 0; j < 8; ++j) {
    const int row = m0 + wave * 16 + j + 8 * half;
    float* yp = Y + (size_t)row * EMBED + o0 + l16;
    yp[0]  = acc[0][j] + bb0;
    yp[16] = acc[1][j] + bb1;
    yp[32] = acc[2][j] + bb2;
    yp[48] = acc[3][j] + bb3;
  }
}

extern "C" void kernel_launch(void* const* d_in, const int* in_sizes, int n_in,
                              void* d_out, int out_size, void* d_ws, size_t ws_size,
                              hipStream_t stream) {
  const float* Vg = (const float*)d_in[0];   // values
  const float* Kg = (const float*)d_in[1];   // keys
  const float* Qg = (const float*)d_in[2];   // query
  const float* Wf = (const float*)d_in[3];   // fc_w [512,512]
  const float* Bf = (const float*)d_in[4];   // fc_b [512]
  float* out  = (float*)d_out;
  float* attn = (float*)d_ws;                // N*S*EMBED f32

  const int N = in_sizes[0] / (S_LEN * EMBED);

  dim3 g1(S_LEN / 128, HEADS, N);
  attn_flash_kernel<<<g1, 128, 0, stream>>>(Vg, Kg, Qg, attn);

  dim3 g2((N * S_LEN) / 64, EMBED / 64);
  fc_out_kernel<<<g2, 128, 0, stream>>>(attn, Wf, Bf, out);
}
